// RNN_3092376453636
// MI455X (gfx1250) — compile-verified, matrix operand order checked by reading, not run on hardware
//
#include <hip/hip_runtime.h>
#include <stdint.h>

#define T_LEN 512
#define BATCH 64
#define IN0   128
#define HID   512
#define GATES 2048   /* 4*HID, rows permuted to (i,f,g,o)-interleaved */
#define OUTN  16
#define NTHR  512    /* 16 waves */

typedef __attribute__((ext_vector_type(16))) __bf16 v16bf;
typedef __attribute__((ext_vector_type(8)))  float  v8f;

union Frag { v16bf v; uint4 q[2]; };
union Acc  { v8f  v; float f[8]; };

__device__ __forceinline__ unsigned short f2bf(float f) {
  unsigned int u = __float_as_uint(f);
  u += 0x7FFFu + ((u >> 16) & 1u);           // round-to-nearest-even
  return (unsigned short)(u >> 16);
}
__device__ __forceinline__ float bf2f(unsigned short s) {
  return __uint_as_float(((unsigned int)s) << 16);
}
__device__ __forceinline__ float sigm(float x) {
  return 1.0f / (1.0f + __expf(-x));
}
__device__ __forceinline__ float ftanh(float x) {
  x = fminf(fmaxf(x, -20.0f), 20.0f);
  float e = __expf(2.0f * x);
  return (e - 1.0f) / (e + 1.0f);
}

// ---------------------------------------------------------------------------
// Weight prep: combined [2][GATES][Kt] bf16, permuted row p <- gate row
// (p&3)*H + (p>>2) (i,f,g,o interleave). Cols [0,H)=W_hh, [H,Kt)=W_ih.
// Fused bias b_ih+b_hh (permuted).
// ---------------------------------------------------------------------------
__global__ void conv_w(const float* __restrict__ Wih, const float* __restrict__ Whh,
                       const float* __restrict__ bih, const float* __restrict__ bhh,
                       int In, int Kt,
                       unsigned short* __restrict__ Wc, float* __restrict__ bias) {
  int i = blockIdx.x * blockDim.x + threadIdx.x;
  int total = 2 * GATES * Kt;
  if (i >= total) return;
  int d   = i / (GATES * Kt);
  int rem = i - d * GATES * Kt;
  int p   = rem / Kt;
  int k   = rem - p * Kt;
  int j   = (p & 3) * HID + (p >> 2);
  float v = (k < HID) ? Whh[((size_t)d * GATES + j) * HID + k]
                      : Wih[((size_t)d * GATES + j) * In + (k - HID)];
  Wc[i] = f2bf(v);
  if (k == 0) bias[d * GATES + p] = bih[d * GATES + j] + bhh[d * GATES + j];
}

__global__ void conv_x(const float* __restrict__ x, unsigned short* __restrict__ xb, int n) {
  int i = blockIdx.x * blockDim.x + threadIdx.x;
  if (i < n) xb[i] = f2bf(x[i]);
}

// ---------------------------------------------------------------------------
// Recurrent scan: blockIdx.x = direction. 16 waves; each wave owns 8 column
// tiles (= 32 hidden units) x all 64 batch rows, processed as 2 passes of
// 4 col tiles with acc[4][4] (128 VGPRs, no spill at 4 waves/SIMD).
// Gates GEMM per step: C[64x2048] = [h | x_t][64xKt] * Wc^T, C init = bias.
// h double-buffered in dynamic LDS; c in global workspace (owner-exclusive).
// ---------------------------------------------------------------------------
__global__ __launch_bounds__(NTHR) void lstm_scan(
    const unsigned short* __restrict__ Wc,    // [2][GATES][Kt] bf16
    const float* __restrict__ bias,           // [2][GATES]
    const unsigned short* __restrict__ xin,   // [T*B][ldin] bf16
    int ldin, int Kt,
    unsigned short* __restrict__ hout,        // [T*B][2*HID] bf16
    float* __restrict__ cws)                  // [2][B*HID] f32
{
  extern __shared__ unsigned short hb2[];     // [2][BATCH*HID] bf16 (128 KB)
  const int dir  = blockIdx.x;
  const int tid  = threadIdx.x;
  const int lane = tid & 31;
  const int wv   = tid >> 5;                  // 0..15
  const int ml   = lane & 15;                 // A/B row within tile, C col
  const int g    = lane >> 4;                 // lane group (K-half select)
  const int xksteps = (Kt - HID) >> 5;        // input-projection K steps

  const unsigned short* __restrict__ Wd = Wc + (size_t)dir * GATES * Kt;
  const float* __restrict__ bd = bias + dir * GATES;
  float* __restrict__ cbuf = cws + (size_t)dir * BATCH * HID;

  const int colbase = wv * 128;               // first gate column of this wave
  // per-lane element offsets (32-bit; bases stay uniform -> saddr loads)
  const int wofs = (colbase + ml) * Kt + 8 * g;
  int hofs[4], xofs[4];
  #pragma unroll
  for (int rt = 0; rt < 4; ++rt) {
    hofs[rt] = (rt * 16 + ml) * HID + 8 * g;
    xofs[rt] = (rt * 16 + ml) * ldin + 8 * g;
  }

  for (int i = tid; i < BATCH * HID; i += NTHR) {
    hb2[i] = 0; hb2[BATCH * HID + i] = 0; cbuf[i] = 0.0f;
  }
  __syncthreads();

  for (int ts = 0; ts < T_LEN; ++ts) {
    const int t_eff = dir ? (T_LEN - 1 - ts) : ts;
    const unsigned short* __restrict__ hprev = hb2 + ((ts + 1) & 1) * (BATCH * HID);
    unsigned short* __restrict__ hcur        = hb2 + (ts & 1) * (BATCH * HID);
    const unsigned short* __restrict__ xrow  = xin + (size_t)t_eff * BATCH * ldin;

    for (int pass = 0; pass < 2; ++pass) {    // 2 passes of 4 column tiles
      Acc acc[4][4];
      #pragma unroll
      for (int c = 0; c < 4; ++c) {
        const float bv = bd[colbase + (pass * 4 + c) * 16 + ml];
        #pragma unroll
        for (int rt = 0; rt < 4; ++rt)
          #pragma unroll
          for (int r = 0; r < 8; ++r) acc[c][rt].f[r] = bv;
      }

      // ---- recurrent part: A = h(t-1) from LDS, K = 0..511 ----
      for (int ks = 0; ks < (HID >> 5); ++ks) {
        Frag a[4];
        #pragma unroll
        for (int rt = 0; rt < 4; ++rt) {
          const unsigned short* p = hprev + hofs[rt] + ks * 32;
          a[rt].q[0] = *(const uint4*)p;
          a[rt].q[1] = *(const uint4*)(p + 16);
        }
        #pragma unroll
        for (int c = 0; c < 4; ++c) {
          const unsigned short* wp = Wd + wofs + (pass * 4 + c) * 16 * Kt + ks * 32;
          Frag b;
          b.q[0] = *(const uint4*)wp;
          b.q[1] = *(const uint4*)(wp + 16);
          #pragma unroll
          for (int rt = 0; rt < 4; ++rt)
            acc[c][rt].v = __builtin_amdgcn_wmma_f32_16x16x32_bf16(
                false, a[rt].v, false, b.v, (short)0, acc[c][rt].v, false, false);
        }
      }
      // ---- input-projection part: A = x_t from global, K = 512..Kt ----
      for (int ks = 0; ks < xksteps; ++ks) {
        Frag a[4];
        #pragma unroll
        for (int rt = 0; rt < 4; ++rt) {
          const unsigned short* p = xrow + xofs[rt] + ks * 32;
          a[rt].q[0] = *(const uint4*)p;
          a[rt].q[1] = *(const uint4*)(p + 16);
        }
        #pragma unroll
        for (int c = 0; c < 4; ++c) {
          const unsigned short* wp =
              Wd + wofs + (pass * 4 + c) * 16 * Kt + (HID >> 5) * 32 + ks * 32;
          Frag b;
          b.q[0] = *(const uint4*)wp;
          b.q[1] = *(const uint4*)(wp + 16);
          #pragma unroll
          for (int rt = 0; rt < 4; ++rt)
            acc[c][rt].v = __builtin_amdgcn_wmma_f32_16x16x32_bf16(
                false, a[rt].v, false, b.v, (short)0, acc[c][rt].v, false, false);
        }
      }

      // ---- elementwise LSTM cell: lane n holds gate (n&3) of unit (col>>2) ----
      #pragma unroll
      for (int c = 0; c < 4; ++c) {
        const int u = (colbase + (pass * 4 + c) * 16 + ml) >> 2; // owner lanes
        #pragma unroll
        for (int rt = 0; rt < 4; ++rt) {
          #pragma unroll
          for (int r = 0; r < 8; ++r) {
            const float gv = acc[c][rt].f[r];
            const int   bl = lane & ~3;
            const float gi = __shfl(gv, bl + 0, 32);
            const float gf = __shfl(gv, bl + 1, 32);
            const float gg = __shfl(gv, bl + 2, 32);
            const float go = __shfl(gv, bl + 3, 32);
            if ((lane & 3) == 0) {
              const int   m  = rt * 16 + r + 8 * g;
              const int   ci = m * HID + u;
              const float cn = sigm(gf) * cbuf[ci] + sigm(gi) * ftanh(gg);
              const float hn = sigm(go) * ftanh(cn);
              cbuf[ci] = cn;
              const unsigned short hb = f2bf(hn);
              hcur[ci] = hb;
              hout[((size_t)t_eff * BATCH + m) * (2 * HID) + dir * HID + u] = hb;
            }
          }
        }
      }
    }
    __syncthreads();   // single barrier per step (double-buffered h)
  }
}

// Final FC: [64, 2H] x [2H, 16] + b
__global__ void fc_kernel(const unsigned short* __restrict__ hlast,
                          const float* __restrict__ w, const float* __restrict__ b,
                          float* __restrict__ out) {
  int tid = blockIdx.x * blockDim.x + threadIdx.x;   // 1024 = 64*16
  int bb = tid >> 4, oo = tid & 15;
  float s = b[oo];
  for (int k = 0; k < 2 * HID; ++k)
    s += bf2f(hlast[bb * (2 * HID) + k]) * w[oo * (2 * HID) + k];
  out[bb * OUTN + oo] = s;
}

extern "C" void kernel_launch(void* const* d_in, const int* in_sizes, int n_in,
                              void* d_out, int out_size, void* d_ws, size_t ws_size,
                              hipStream_t stream) {
  (void)in_sizes; (void)n_in; (void)out_size; (void)ws_size;
  const float* x    = (const float*)d_in[0];
  const float* Wih0 = (const float*)d_in[1];
  const float* Whh0 = (const float*)d_in[2];
  const float* bih0 = (const float*)d_in[3];
  const float* bhh0 = (const float*)d_in[4];
  const float* Wih1 = (const float*)d_in[5];
  const float* Whh1 = (const float*)d_in[6];
  const float* bih1 = (const float*)d_in[7];
  const float* bhh1 = (const float*)d_in[8];
  const float* fcw  = (const float*)d_in[9];
  const float* fcb  = (const float*)d_in[10];
  float* out = (float*)d_out;

  char* ws = (char*)d_ws;
  size_t off = 0;
  auto alloc = [&](size_t bytes) -> char* {
    char* p = ws + off;
    off = (off + bytes + 255) & ~(size_t)255;
    return p;
  };
  const int Kt0 = HID + IN0;        // 640
  const int Kt1 = HID + 2 * HID;    // 1536
  unsigned short* Xb  = (unsigned short*)alloc((size_t)T_LEN * BATCH * IN0 * 2);
  unsigned short* Wc0 = (unsigned short*)alloc((size_t)2 * GATES * Kt0 * 2);
  float*          Bs0 = (float*)         alloc((size_t)2 * GATES * 4);
  unsigned short* Wc1 = (unsigned short*)alloc((size_t)2 * GATES * Kt1 * 2);
  float*          Bs1 = (float*)         alloc((size_t)2 * GATES * 4);
  unsigned short* O0  = (unsigned short*)alloc((size_t)T_LEN * BATCH * 2 * HID * 2);
  unsigned short* O1  = (unsigned short*)alloc((size_t)T_LEN * BATCH * 2 * HID * 2);
  float*          C0  = (float*)         alloc((size_t)2 * BATCH * HID * 4);
  float*          C1  = (float*)         alloc((size_t)2 * BATCH * HID * 4);

  int n;
  n = T_LEN * BATCH * IN0;
  conv_x<<<(n + 255) / 256, 256, 0, stream>>>(x, Xb, n);
  n = 2 * GATES * Kt0;
  conv_w<<<(n + 255) / 256, 256, 0, stream>>>(Wih0, Whh0, bih0, bhh0, IN0, Kt0, Wc0, Bs0);
  n = 2 * GATES * Kt1;
  conv_w<<<(n + 255) / 256, 256, 0, stream>>>(Wih1, Whh1, bih1, bhh1, 2 * HID, Kt1, Wc1, Bs1);

  const size_t shmem = (size_t)2 * BATCH * HID * sizeof(unsigned short); // 128 KB
  lstm_scan<<<2, NTHR, shmem, stream>>>(Wc0, Bs0, Xb, IN0,     Kt0, O0, C0);
  lstm_scan<<<2, NTHR, shmem, stream>>>(Wc1, Bs1, O0, 2 * HID, Kt1, O1, C1);

  fc_kernel<<<1, 1024, 0, stream>>>(O1 + (size_t)(T_LEN - 1) * BATCH * 2 * HID,
                                    fcw, fcb, out);
}
